// GeometricCosineAttention_369367187597
// MI455X (gfx1250) — matrix operand order repaired and checked by reading, not compiled, not run on hardware
//
#include <hip/hip_runtime.h>
#include <hip/hip_bf16.h>

// ---- problem constants (match reference) ----
#define BB 2
#define EE 2048
#define CC 512
#define HH 8
#define DD 64
#define LN_EPS 1e-5f

#define USE_TDM 1   // Tensor Data Mover path for the attention K-tile

typedef __attribute__((ext_vector_type(16))) __bf16 v16bf;
typedef __attribute__((ext_vector_type(8)))  float  v8f;
typedef __attribute__((ext_vector_type(4)))  unsigned int v4u;
typedef __attribute__((ext_vector_type(8)))  int    v8i_t;
typedef __attribute__((ext_vector_type(4)))  int    v4i_t;
typedef __bf16 bf16;

static __device__ __forceinline__ bf16 tobf(float f) { return (bf16)f; }

// A-fragment (16x32 bf16): lane holds row m = lane&15; K values:
//   e=0..7  -> K = kb + e        (kb = 8*(lane>>4))
//   e=8..15 -> K = 16 + kb + (e-8)
static __device__ __forceinline__ v16bf load_a_frag(const bf16* rowptr, int lane) {
  const int kb = (lane >> 4) * 8;
  v16bf a;
#pragma unroll
  for (int e = 0; e < 8; ++e) a[e] = rowptr[kb + e];
#pragma unroll
  for (int e = 0; e < 8; ++e) a[8 + e] = rowptr[16 + kb + e];
  return a;
}

// Same but converting f32 -> bf16 on load (for fp32 activations).
static __device__ __forceinline__ v16bf load_a_frag_f32(const float* rowptr, int lane) {
  const int kb = (lane >> 4) * 8;
  v16bf a;
#pragma unroll
  for (int e = 0; e < 8; ++e) a[e] = tobf(rowptr[kb + e]);
#pragma unroll
  for (int e = 0; e < 8; ++e) a[8 + e] = tobf(rowptr[16 + kb + e]);
  return a;
}

// B-fragment (32x16 bf16): lane holds column N = lane&15 (p points at K=0 of it);
//   K order same as A-fragment, K-row stride = stride.
static __device__ __forceinline__ v16bf load_b_frag(const bf16* p, int stride, int lane) {
  const int kb = (lane >> 4) * 8;
  v16bf b;
#pragma unroll
  for (int e = 0; e < 8; ++e) b[e] = p[(size_t)(kb + e) * stride];
#pragma unroll
  for (int e = 0; e < 8; ++e) b[8 + e] = p[(size_t)(16 + kb + e) * stride];
  return b;
}

#if USE_TDM
// TDM 2D tile load: 64 rows x 32 bf16, global row stride EE elements,
// LDS row stride 40 bf16 (pad 16B after every 64B row: interval=16DW, amount=4DW).
// D# packing per CDNA5 ISA.
// This toolchain exposes the 6-arg builtin:
//   (uint32x4 g0, int32x8 g1, int32x4 g2, int32x4 g3, int32x8, i32 cpol)
static __device__ __forceinline__ void tdm_load_kT(const bf16* gsrc, bf16* ldst) {
  const unsigned long long ga = (unsigned long long)(uintptr_t)gsrc;
  const unsigned int la = (unsigned int)(uintptr_t)ldst;  // low 32b of flat = LDS offset
  v4u g0 = { 1u,                                   // count=1, user mode
             la,                                   // lds_addr
             (unsigned int)(ga & 0xFFFFFFFFu),     // global_addr[31:0]
             (unsigned int)((ga >> 32) & 0x01FFFFFFu) | (2u << 30) }; // addr[56:32] | type=2
  v8i_t g1 = { (int)0x06D10000u,   // wg_mask=0 | data_size=1(2B)<<16 | pad_en<<20
                                   // | pad_interval=3(16DW)<<22 | pad_amount=3(4DW)<<25
               (int)(32u << 16),   // tensor_dim0 = 32   (bits[63:48] = lo16)
               (int)(64u << 16),   // tensor_dim1 = 64   (bits[95:80] = lo16)
               (int)(32u << 16),   // tile_dim0   = 32   (bits[127:112])
               64,                 // tile_dim1   = 64   (bits[143:128]); tile_dim2=0
               (int)EE,            // tensor_dim0_stride lo32 = 2048
               0, 0 };
  v4i_t z4 = { 0, 0, 0, 0 };
  v8i_t z8 = { 0, 0, 0, 0, 0, 0, 0, 0 };
  __builtin_amdgcn_tensor_load_to_lds(g0, g1, z4, z4, z8, 0);
}
#endif

// =====================================================================
// Kernel 0: transpose + convert weights: WT[c][o] = (bf16)W[o][c]
// grid = (C*C/256, 4), block = 256
// =====================================================================
__global__ __launch_bounds__(256)
void transpose_w_kernel(const float* __restrict__ Wq,
                        const float* __restrict__ Wk,
                        const float* __restrict__ Wv,
                        const float* __restrict__ Wp,
                        bf16* __restrict__ WqT, bf16* __restrict__ WkT,
                        bf16* __restrict__ WvT, bf16* __restrict__ WpT) {
  const int z = blockIdx.y;
  const float* W = (z == 0) ? Wq : (z == 1) ? Wk : (z == 2) ? Wv : Wp;
  bf16* WT = (z == 0) ? WqT : (z == 1) ? WkT : (z == 2) ? WvT : WpT;
  const int idx = blockIdx.x * 256 + threadIdx.x;  // < C*C
  const int c = idx >> 9, o = idx & (CC - 1);
  WT[(size_t)c * CC + o] = tobf(W[(size_t)o * CC + c]);  // write coalesced in o
}

// =====================================================================
// Kernel 1: q/k/v projections, LDS-free.  y = x @ W.T
//   grid = (B*E/64, C/64, 3), block = 128 (4 waves)
//   which==0 -> q [B,H,E,D]; which==1 -> kT [B,H,D,E]; which==2 -> v [B,H,E,D]
// =====================================================================
__global__ __launch_bounds__(128)
void proj_qkv_kernel(const float* __restrict__ x,
                     const bf16* __restrict__ WqT,
                     const bf16* __restrict__ WkT,
                     const bf16* __restrict__ WvT,
                     bf16* __restrict__ qb,
                     bf16* __restrict__ kTb,
                     bf16* __restrict__ vb) {
  const int which = blockIdx.z;
  const bf16* WT = (which == 0) ? WqT : (which == 1) ? WkT : WvT;
  bf16* outp = (which == 0) ? qb : (which == 1) ? kTb : vb;

  const int m0 = blockIdx.x * 64;
  const int n0 = blockIdx.y * 64;
  const int lane = threadIdx.x & 31;
  const int wv = threadIdx.x >> 5;

  const float* xrow = x + (size_t)(m0 + wv * 16 + (lane & 15)) * CC;

  v8f acc[4] = {};
  for (int k0 = 0; k0 < CC; k0 += 32) {
    v16bf afrag = load_a_frag_f32(xrow + k0, lane);
    const bf16* wrow = WT + (size_t)k0 * CC + n0 + (lane & 15);  // lane = N-col
#pragma unroll
    for (int t = 0; t < 4; ++t) {
      v16bf bfrag = load_b_frag(wrow + t * 16, CC, lane);
      acc[t] = __builtin_amdgcn_wmma_f32_16x16x32_bf16(
          false, afrag, false, bfrag, (short)0, acc[t], false, false);
    }
  }

  const int ncol = lane & 15;
#pragma unroll
  for (int t = 0; t < 4; ++t) {
    const int o = n0 + t * 16 + ncol;
    const int h = o >> 6, d = o & 63;
#pragma unroll
    for (int vg = 0; vg < 8; ++vg) {
      const int gm = m0 + wv * 16 + vg + 8 * (lane >> 4);
      const int b = gm >> 11, e = gm & (EE - 1);
      size_t idx;
      if (which == 1)
        idx = (((size_t)(b * HH + h) * DD + d) * EE + e);
      else
        idx = (((size_t)(b * HH + h) * EE + e) * DD + d);
      outp[idx] = tobf(acc[t][vg]);
    }
  }
}

// =====================================================================
// Kernel 2: fused attention per (b,h):
//   s = q@k^T ; s = (s*mask+1)*0.5 ; out += s@v   (scores stay on-chip)
// grid = (E/64, B*H), block = 128 (4 waves, 16 q-rows each).
// K-tiles double-buffered via TDM async DMA.
// =====================================================================
__global__ __launch_bounds__(128)
void attn_kernel(const bf16* __restrict__ qb,
                 const bf16* __restrict__ kTb,
                 const bf16* __restrict__ vb,
                 const float* __restrict__ adj,
                 bf16* __restrict__ aout) {
  const int bh = blockIdx.y;
  const int b = bh / HH, h = bh % HH;
  const int m0 = blockIdx.x * 64;
  const int tid = threadIdx.x;
  const int lane = tid & 31;
  const int wv = tid >> 5;

  __shared__ bf16 ldsK[2][64][40];     // double-buffered kT tile [d][token]
  __shared__ bf16 ldsS[4][16][40];     // per-wave score tile [m][token(32)]

  const bf16* kbase = kTb + (size_t)bh * DD * EE;  // [d][e], row stride EE

  // q A-fragments for this wave's 16 rows (K = D = 64 -> 2 fragments), hoisted
  const bf16* qrow = qb + ((size_t)bh * EE + m0 + wv * 16 + (lane & 15)) * DD;
  const v16bf aq0 = load_a_frag(qrow, lane);
  const v16bf aq1 = load_a_frag(qrow + 32, lane);

  v8f oacc[4] = {};

#if USE_TDM
  if (tid < 32) tdm_load_kT(kbase, &ldsK[0][0][0]);   // prologue: tile 0
#endif

  for (int it = 0; it < EE / 32; ++it) {
    const int kc = it * 32;
    const int cur = it & 1;

#if USE_TDM
    if (tid < 32) {
      if (it + 1 < EE / 32) {
        // buffer cur^1 was last read in iteration it-1; the trailing
        // __syncthreads() of that iteration makes this overwrite safe.
        tdm_load_kT(kbase + (kc + 32), &ldsK[cur ^ 1][0][0]);
        __builtin_amdgcn_s_wait_tensorcnt(1);  // tile `it` done, next in flight
      } else {
        __builtin_amdgcn_s_wait_tensorcnt(0);
      }
    }
#else
    for (int i = tid; i < 64 * 32; i += 128) {
      int d = i >> 5, n = i & 31;
      ldsK[cur][d][n] = kbase[(size_t)d * EE + kc + n];
    }
#endif
    __syncthreads();  // ldsK[cur] visible to all waves

    if (kc + 32 < EE)  // prefetch next mask row-chunk (L2-resident)
      __builtin_prefetch(adj + (size_t)(m0 + wv * 16 + (lane & 15)) * EE + kc + 32, 0, 1);

    // ---- scores: 2 N-tiles of 16 key-tokens, K = 64 (2 WMMA each) ----
#pragma unroll
    for (int t = 0; t < 2; ++t) {
      v8f s = {};
      v16bf bk0 = load_b_frag(&ldsK[cur][0][t * 16 + (lane & 15)], 40, lane);   // d = 0..31
      v16bf bk1 = load_b_frag(&ldsK[cur][32][t * 16 + (lane & 15)], 40, lane);  // d = 32..63
      s = __builtin_amdgcn_wmma_f32_16x16x32_bf16(
          false, aq0, false, bk0, (short)0, s, false, false);
      s = __builtin_amdgcn_wmma_f32_16x16x32_bf16(
          false, aq1, false, bk1, (short)0, s, false, false);

      const int n = t * 16 + (lane & 15);
      const int gq_base = m0 + wv * 16 + 8 * (lane >> 4);
#pragma unroll
      for (int vg = 0; vg < 8; ++vg) {
        const int gq = gq_base + vg;
        const float mval = adj[(size_t)gq * EE + kc + n];
        const float sv = (s[vg] * mval + 1.0f) * 0.5f;
        ldsS[wv][vg + 8 * (lane >> 4)][n] = tobf(sv);
      }
    }

    // ldsS write->read is intra-wave; LDS ops from one wave complete in
    // order, so only a scheduling barrier is needed (no block barrier).
    __builtin_amdgcn_wave_barrier();

    // ---- out += s @ v : A = score frag (16x32), B = v (32 tok x 64 d) ----
    v16bf as = load_a_frag(&ldsS[wv][lane & 15][0], lane);
    const bf16* vrow = vb + ((size_t)bh * EE + kc) * DD + (lane & 15);  // lane = N-col
#pragma unroll
    for (int t = 0; t < 4; ++t) {
      v16bf bv = load_b_frag(vrow + t * 16, DD, lane);
      oacc[t] = __builtin_amdgcn_wmma_f32_16x16x32_bf16(
          false, as, false, bv, (short)0, oacc[t], false, false);
    }
    __syncthreads();  // all waves done with ldsK[cur] before it is re-filled
  }

  // store heads re-interleaved: aout[B,E,C] with channel = h*64 + d
#pragma unroll
  for (int t = 0; t < 4; ++t) {
    const int d = t * 16 + (lane & 15);
#pragma unroll
    for (int vg = 0; vg < 8; ++vg) {
      const int e = m0 + wv * 16 + vg + 8 * (lane >> 4);
      aout[((size_t)b * EE + e) * CC + h * DD + d] = tobf(oacc[t][vg]);
    }
  }
}

// =====================================================================
// Kernel 3: final projection, LDS-free.  proj = aout @ Wp.T + bp (f32)
// grid = (B*E/64, C/64), block = 128
// =====================================================================
__global__ __launch_bounds__(128)
void proj_out_kernel(const bf16* __restrict__ aout,
                     const bf16* __restrict__ WpT,
                     const float* __restrict__ bp,
                     float* __restrict__ proj) {
  const int m0 = blockIdx.x * 64;
  const int n0 = blockIdx.y * 64;
  const int lane = threadIdx.x & 31;
  const int wv = threadIdx.x >> 5;

  const bf16* arow = aout + (size_t)(m0 + wv * 16 + (lane & 15)) * CC;

  v8f acc[4] = {};
  for (int k0 = 0; k0 < CC; k0 += 32) {
    v16bf afrag = load_a_frag(arow + k0, lane);
    const bf16* wrow = WpT + (size_t)k0 * CC + n0 + (lane & 15);
#pragma unroll
    for (int t = 0; t < 4; ++t) {
      v16bf bfrag = load_b_frag(wrow + t * 16, CC, lane);
      acc[t] = __builtin_amdgcn_wmma_f32_16x16x32_bf16(
          false, afrag, false, bfrag, (short)0, acc[t], false, false);
    }
  }

  const int ncol = lane & 15;
#pragma unroll
  for (int t = 0; t < 4; ++t) {
    const int o = n0 + t * 16 + ncol;
    const float bias = bp[o];
#pragma unroll
    for (int vg = 0; vg < 8; ++vg) {
      const int gm = m0 + wv * 16 + vg + 8 * (lane >> 4);
      proj[(size_t)gm * CC + o] = acc[t][vg] + bias;
    }
  }
}

// =====================================================================
// Kernel 4: LayerNorm + gated residual:  out = x + gamma * LN(proj)
// grid = B*E, block = 128
// =====================================================================
__global__ __launch_bounds__(128)
void ln_residual_kernel(const float* __restrict__ proj,
                        const float* __restrict__ x,
                        const float* __restrict__ lnw,
                        const float* __restrict__ lnb,
                        const float* __restrict__ gamma,
                        float* __restrict__ out) {
  const int row = blockIdx.x;
  const int tid = threadIdx.x;
  const float* p = proj + (size_t)row * CC;

  float v[4];
  float s = 0.f, s2 = 0.f;
#pragma unroll
  for (int i = 0; i < 4; ++i) {
    v[i] = p[tid + i * 128];
    s += v[i];
    s2 += v[i] * v[i];
  }

  __shared__ float rs[128], rs2[128];
  rs[tid] = s;
  rs2[tid] = s2;
  __syncthreads();
  for (int off = 64; off > 0; off >>= 1) {
    if (tid < off) {
      rs[tid] += rs[tid + off];
      rs2[tid] += rs2[tid + off];
    }
    __syncthreads();
  }
  const float mu = rs[0] * (1.0f / CC);
  const float var = rs2[0] * (1.0f / CC) - mu * mu;
  const float inv = rsqrtf(var + LN_EPS);
  const float g = gamma[0];

#pragma unroll
  for (int i = 0; i < 4; ++i) {
    const int c = tid + i * 128;
    const float ln = (v[i] - mu) * inv * lnw[c] + lnb[c];
    out[(size_t)row * CC + c] = x[(size_t)row * CC + c] + g * ln;
  }
}

// =====================================================================
extern "C" void kernel_launch(void* const* d_in, const int* in_sizes, int n_in,
                              void* d_out, int out_size, void* d_ws, size_t ws_size,
                              hipStream_t stream) {
  const float* x    = (const float*)d_in[0];
  const float* adj  = (const float*)d_in[1];
  const float* Wq   = (const float*)d_in[2];
  const float* Wk   = (const float*)d_in[3];
  const float* Wv   = (const float*)d_in[4];
  const float* Wp   = (const float*)d_in[5];
  const float* bp   = (const float*)d_in[6];
  const float* lnw  = (const float*)d_in[7];
  const float* lnb  = (const float*)d_in[8];
  const float* gam  = (const float*)d_in[9];
  float* out = (float*)d_out;

  // workspace carve-up
  const size_t NTOK  = (size_t)BB * EE;            // 4096
  const size_t HEDSZ = (size_t)BB * HH * EE * DD;  // 2,097,152 bf16 each
  const size_t WSZ   = (size_t)CC * CC;            // 262,144 bf16 each
  bf16* qb   = (bf16*)d_ws;
  bf16* kTb  = qb + HEDSZ;
  bf16* vb   = kTb + HEDSZ;
  bf16* aout = vb + HEDSZ;                   // NTOK*CC bf16
  bf16* WqT  = aout + NTOK * CC;
  bf16* WkT  = WqT + WSZ;
  bf16* WvT  = WkT + WSZ;
  bf16* WpT  = WvT + WSZ;
  float* proj = (float*)(WpT + WSZ);         // NTOK*CC f32
  (void)in_sizes; (void)n_in; (void)out_size; (void)ws_size;

  transpose_w_kernel<<<dim3(WSZ / 256, 4), 256, 0, stream>>>(
      Wq, Wk, Wv, Wp, WqT, WkT, WvT, WpT);
  proj_qkv_kernel<<<dim3(NTOK / 64, CC / 64, 3), 128, 0, stream>>>(
      x, WqT, WkT, WvT, qb, kTb, vb);
  attn_kernel<<<dim3(EE / 64, BB * HH), 128, 0, stream>>>(
      qb, kTb, vb, adj, aout);
  proj_out_kernel<<<dim3(NTOK / 64, CC / 64), 128, 0, stream>>>(
      aout, WpT, bp, proj);
  ln_residual_kernel<<<NTOK, 128, 0, stream>>>(
      proj, x, lnw, lnb, gam, out);
}
